// HieTree_9878424781091
// MI455X (gfx1250) — matrix operand order repaired and checked by reading, not compiled, not run on hardware
//
#include <hip/hip_runtime.h>
#include <hip/hip_bf16.h>

// ---------------------------------------------------------------------------
// MI455X (gfx1250) implementation.
// HBM-bound problem: 512 MB of fp32 weights streamed once (~22 us floor at
// 23.3 TB/s). GEMMs use V_WMMA_F32_16X16X32_BF16 with an on-the-fly
// bf16(hi)+bf16(lo) split of the fp32 weights (3 WMMAs per K-step) to keep
// near-fp32 accuracy while staying memory-bound. Weights are loaded with
// non-temporal hints so the touch-once 512 MB stream does not evict the
// L2-resident activation / partial-sum working set (~20 MB, 192 MB L2).
// Split-K factor is chosen from ws_size (16 -> 8 -> 4) to keep >=2k waves
// in flight for bandwidth saturation.
// ---------------------------------------------------------------------------

#define DD      4096
#define HALF_D  2048
#define NROWS   54
#define MPAD    64
#define LAYERS  4

typedef __attribute__((ext_vector_type(16))) __bf16 v16bf;
typedef __attribute__((ext_vector_type(8)))  __bf16 v8bf;
typedef __attribute__((ext_vector_type(8)))  float  v8f;
typedef __attribute__((ext_vector_type(4)))  float  v4f;

static __device__ __forceinline__ v8f zero_v8f() {
    v8f z;
#pragma unroll
    for (int i = 0; i < 8; ++i) z[i] = 0.0f;
    return z;
}

// Non-temporal b128 load for the touch-once weight stream.
static __device__ __forceinline__ v4f ldnt4(const float* p) {
    return __builtin_nontemporal_load((const v4f*)p);
}

// ---------------------------------------------------------------------------
// x (54x4096 fp32, padded to 64 rows with zeros) initialization from inputs.
// ---------------------------------------------------------------------------
__global__ void k_init_x(const float* __restrict__ ce, float* __restrict__ x) {
    int idx = blockIdx.x * blockDim.x + threadIdx.x;  // 64*4096 threads
    int r = idx >> 12;
    x[idx] = (r < NROWS) ? ce[idx] : 0.0f;
}

// ---------------------------------------------------------------------------
// Split fp32 activations into bf16 hi + bf16 lo (residual) buffers.
// ---------------------------------------------------------------------------
__global__ void k_prep(const float* __restrict__ x,
                       __bf16* __restrict__ xh, __bf16* __restrict__ xl) {
    int idx = blockIdx.x * blockDim.x + threadIdx.x;
    float f = x[idx];
    __bf16 hb = (__bf16)f;
    xh[idx] = hb;
    xl[idx] = (__bf16)(f - (float)hb);
}

// ---------------------------------------------------------------------------
// Split-K GEMM:  hpart[k][n][m] = sum_{K in slice k} W[n][K] * x[m][K]
// A operand = W tile (16 n-rows x 32 K, bf16 hi/lo from fp32 stream, NT loads)
// B operand = x^T tile (32 K x 16 m), read from pre-split bf16 buffers (L2).
// grid = (64, SK), block = 128 (4 waves); each wave owns a 16-row n tile
// and all 4 m-tiles (M padded to 64).
// ---------------------------------------------------------------------------
template <int SK>
__global__ void __launch_bounds__(128)
k_gemm(const float* __restrict__ W,
       const __bf16* __restrict__ xh, const __bf16* __restrict__ xl,
       float* __restrict__ hpart) {
    constexpr int KSLICE = DD / SK;
    const int lane  = threadIdx.x & 31;
    const int wave  = threadIdx.x >> 5;
    const int n0    = blockIdx.x * 64 + wave * 16;
    const int kbase = blockIdx.y * KSLICE;

    // A-fragment layout (bf16 16x32): lanes 0-15 own K {0..7,16..23} of row
    // M=lane; lanes 16-31 own K {8..15,24..31} of row M=lane-16.
    const int  rowA   = n0 + (lane & 15);
    const int  khalfA = (lane >> 4) * 8;
    const int  koffB  = (lane >> 4) * 8;    // B: lanes 0-15 K{0..7,16..23}, 16-31 K{8..15,24..31}
    const int  mB     = lane & 15;
    const float* wp   = W + (size_t)rowA * DD + kbase + khalfA;

    v8f acc[4];
#pragma unroll
    for (int mt = 0; mt < 4; ++mt) acc[mt] = zero_v8f();

    for (int k = 0; k < KSLICE; k += 32) {
        v4f a0 = ldnt4(wp + k);
        v4f a1 = ldnt4(wp + k + 4);
        v4f a2 = ldnt4(wp + k + 16);
        v4f a3 = ldnt4(wp + k + 20);
        __builtin_prefetch(wp + k + 512, 0, 1);  // global_prefetch_b8, 2KB ahead

        float fa[16];
#pragma unroll
        for (int i = 0; i < 4; ++i) {
            fa[i] = a0[i]; fa[4 + i] = a1[i]; fa[8 + i] = a2[i]; fa[12 + i] = a3[i];
        }
        v16bf ah, al;
#pragma unroll
        for (int i = 0; i < 16; ++i) {
            float f = fa[i];
            __bf16 hb = (__bf16)f;
            ah[i] = hb;
            al[i] = (__bf16)(f - (float)hb);
        }

#pragma unroll
        for (int mt = 0; mt < 4; ++mt) {
            const size_t boff = (size_t)(mt * 16 + mB) * DD + kbase + k + koffB;
            v8bf bh0 = *(const v8bf*)(xh + boff);
            v8bf bh1 = *(const v8bf*)(xh + boff + 16);
            v8bf bl0 = *(const v8bf*)(xl + boff);
            v8bf bl1 = *(const v8bf*)(xl + boff + 16);
            v16bf bh = __builtin_shufflevector(bh0, bh1,
                0, 1, 2, 3, 4, 5, 6, 7, 8, 9, 10, 11, 12, 13, 14, 15);
            v16bf bl = __builtin_shufflevector(bl0, bl1,
                0, 1, 2, 3, 4, 5, 6, 7, 8, 9, 10, 11, 12, 13, 14, 15);
            // bf16x3 product: hi*hi + hi*lo + lo*hi  (fp32 accumulate)
            acc[mt] = __builtin_amdgcn_wmma_f32_16x16x32_bf16(
                false, ah, false, bh, (short)0, acc[mt], false, false);
            acc[mt] = __builtin_amdgcn_wmma_f32_16x16x32_bf16(
                false, ah, false, bl, (short)0, acc[mt], false, false);
            acc[mt] = __builtin_amdgcn_wmma_f32_16x16x32_bf16(
                false, al, false, bh, (short)0, acc[mt], false, false);
        }
    }

    // D layout: vgpr j, lanes 0-15 -> (M=j, N=lane); lanes 16-31 -> (M=8+j).
    float* out = hpart + (size_t)blockIdx.y * DD * MPAD;
    const int nadd = (lane >> 4) * 8;
#pragma unroll
    for (int mt = 0; mt < 4; ++mt) {
        const int m = mt * 16 + mB;
#pragma unroll
        for (int j = 0; j < 8; ++j) {
            out[(size_t)(n0 + j + nadd) * MPAD + m] = acc[mt][j];
        }
    }
}

// ---------------------------------------------------------------------------
// Deterministic split-K reduce + transpose: h[m][n] = sum_k hpart[k][n][m]
// ---------------------------------------------------------------------------
template <int SK>
__global__ void __launch_bounds__(256)
k_reduce(const float* __restrict__ hpart, float* __restrict__ h) {
    __shared__ float tile[16][65];
    const int n0 = blockIdx.x * 16;
    const int t  = threadIdx.x;
#pragma unroll
    for (int rep = 0; rep < 4; ++rep) {
        int e  = t + rep * 256;
        int nr = e >> 6;
        int m  = e & 63;
        size_t base = (size_t)(n0 + nr) * MPAD + m;
        float s = 0.0f;
#pragma unroll
        for (int k = 0; k < SK; ++k) s += hpart[base + (size_t)k * DD * MPAD];
        tile[nr][m] = s;
    }
    __syncthreads();
    int m    = t >> 2;
    int nrel = (t & 3) * 4;
    float4 v;
    v.x = tile[nrel + 0][m];
    v.y = tile[nrel + 1][m];
    v.z = tile[nrel + 2][m];
    v.w = tile[nrel + 3][m];
    *(float4*)(h + (size_t)m * DD + n0 + nrel) = v;
}

// ---------------------------------------------------------------------------
// Attention collapsed to coefficients: all 324 length-4096 dot products via
// wave shfl_xor reductions, then scalar softmax algebra producing the 18x54
// mixing matrix cf: new_row[r] = sum_s cf[r][s] * h[s].
// Single block of 1024 threads (32 waves).
// ---------------------------------------------------------------------------
__global__ void __launch_bounds__(1024)
k_attn_coef(const float* __restrict__ h,
            const float* __restrict__ att_if, const float* __restrict__ att_fd,
            const float* __restrict__ att_dr,
            const int* __restrict__ pf, const int* __restrict__ pidx,
            float* __restrict__ coef) {
    __shared__ float dots[54][6];   // per row: u_if,v_if,u_fd,v_fd,u_dr,v_dr
    __shared__ float cf[18][54];
    const int lane = threadIdx.x & 31;
    const int wave = threadIdx.x >> 5;

    for (int task = wave; task < 54 * 6; task += 32) {
        int r = task / 6, j = task % 6;
        const float* wv = (j < 2 ? att_if : (j < 4 ? att_fd : att_dr)) + (j & 1) * DD;
        const float* hr = h + (size_t)r * DD;
        float p = 0.0f;
        for (int d = lane; d < DD; d += 32) p += hr[d] * wv[d];
#pragma unroll
        for (int o = 16; o; o >>= 1) p += __shfl_xor(p, o, 32);
        if (lane == 0) dots[r][j] = p;
    }
    __syncthreads();

    if (threadIdx.x == 0) {
        for (int r = 0; r < 18; ++r)
            for (int s = 0; s < 54; ++s) cf[r][s] = 0.0f;
        // ---- Level 1: fac (rows 6..17) over ideo (rows 18..53)
        for (int p = 0; p < 12; ++p) {
            float sself = dots[6 + p][0] + dots[6 + p][1];
            sself = sself > 0.f ? sself : 0.01f * sself;
            float scores[36]; int cids[36]; int nc = 0;
            float mx = sself;
            for (int c = 0; c < 36; ++c) if (pidx[c] == p) {
                float s = dots[6 + p][0] + dots[18 + c][1];
                s = s > 0.f ? s : 0.01f * s;
                scores[nc] = s; cids[nc] = c; ++nc;
                if (s > mx) mx = s;
            }
            float e0 = __expf(sself - mx), den = e0;
            float ec[36];
            for (int t = 0; t < nc; ++t) { ec[t] = __expf(scores[t] - mx); den += ec[t]; }
            cf[6 + p][6 + p] = e0 / den;
            for (int t = 0; t < nc; ++t) cf[6 + p][18 + cids[t]] = ec[t] / den;
        }
        // ---- Level 2: dom (rows 1..5) over fac_new
        for (int p = 0; p < 5; ++p) {
            float sself = dots[1 + p][2] + dots[1 + p][3];
            sself = sself > 0.f ? sself : 0.01f * sself;
            float scores[12]; int cids[12]; int nc = 0;
            float mx = sself;
            for (int q = 0; q < 12; ++q) if (pf[q] == p) {
                float vnew = 0.0f;
                for (int s = 0; s < 54; ++s) vnew += cf[6 + q][s] * dots[s][3];
                float s = dots[1 + p][2] + vnew;
                s = s > 0.f ? s : 0.01f * s;
                scores[nc] = s; cids[nc] = q; ++nc;
                if (s > mx) mx = s;
            }
            float e0 = __expf(sself - mx), den = e0;
            float ec[12];
            for (int t = 0; t < nc; ++t) { ec[t] = __expf(scores[t] - mx); den += ec[t]; }
            float a0 = e0 / den;
            cf[1 + p][1 + p] += a0;
            for (int t = 0; t < nc; ++t) {
                float a = ec[t] / den;
                for (int s = 0; s < 54; ++s) cf[1 + p][s] += a * cf[6 + cids[t]][s];
            }
        }
        // ---- Level 3: root (row 0) over dom_new (all 5 children)
        {
            float sself = dots[0][4] + dots[0][5];
            sself = sself > 0.f ? sself : 0.01f * sself;
            float scores[5];
            float mx = sself;
            for (int p = 0; p < 5; ++p) {
                float vnew = 0.0f;
                for (int s = 0; s < 54; ++s) vnew += cf[1 + p][s] * dots[s][5];
                float s = dots[0][4] + vnew;
                s = s > 0.f ? s : 0.01f * s;
                scores[p] = s;
                if (s > mx) mx = s;
            }
            float e0 = __expf(sself - mx), den = e0;
            float ec[5];
            for (int p = 0; p < 5; ++p) { ec[p] = __expf(scores[p] - mx); den += ec[p]; }
            cf[0][0] += e0 / den;
            for (int p = 0; p < 5; ++p) {
                float a = ec[p] / den;
                for (int s = 0; s < 54; ++s) cf[0][s] += a * cf[1 + p][s];
            }
        }
    }
    __syncthreads();
    for (int e = threadIdx.x; e < 18 * 54; e += 1024) coef[e] = cf[e / 54][e % 54];
}

// ---------------------------------------------------------------------------
// Apply attention coefficients: x[r] = sum_s cf[r][s]*h[s] (r<18), copy ideo.
// ---------------------------------------------------------------------------
__global__ void __launch_bounds__(256)
k_attn_apply(const float* __restrict__ h, const float* __restrict__ coef,
             float* __restrict__ x) {
    int r = blockIdx.x >> 4;                       // 16 blocks per row
    int d = ((blockIdx.x & 15) << 8) + threadIdx.x;
    if (r < 18) {
        const float* hd = h + d;
        float s = 0.0f;
#pragma unroll 6
        for (int q = 0; q < 54; ++q) s += coef[r * 54 + q] * hd[(size_t)q * DD];
        x[(size_t)r * DD + d] = s;
    } else {
        x[(size_t)r * DD + d] = h[(size_t)r * DD + d];
    }
}

// ---------------------------------------------------------------------------
// Complex-rotation mix stage (purely columnwise): 2048 threads, one per d.
// ---------------------------------------------------------------------------
__global__ void __launch_bounds__(256)
k_mix(const float* __restrict__ h, const float* __restrict__ edge,
      const int* __restrict__ pf, const int* __restrict__ pidx,
      float* __restrict__ x) {
    int d = blockIdx.x * 256 + threadIdx.x;        // 0..2047
    float er0 = __cosf(edge[d]),              ei0 = __sinf(edge[d]);
    float er1 = __cosf(edge[HALF_D + d]),     ei1 = __sinf(edge[HALF_D + d]);
    float er2 = __cosf(edge[2 * HALF_D + d]), ei2 = __sinf(edge[2 * HALF_D + d]);

    float rr = h[d], ri = h[HALF_D + d];
    x[d] = rr; x[HALF_D + d] = ri;                 // row 0 passes through

    float dr[5], di[5];
#pragma unroll
    for (int p = 0; p < 5; ++p) {
        dr[p] = h[(size_t)(1 + p) * DD + d]          + (rr * er0 - ri * ei0);
        di[p] = h[(size_t)(1 + p) * DD + HALF_D + d] + (rr * ei0 + ri * er0);
        x[(size_t)(1 + p) * DD + d]          = dr[p] * 0.5f;
        x[(size_t)(1 + p) * DD + HALF_D + d] = di[p] * 0.5f;
    }
    float fr[12], fi[12];
#pragma unroll
    for (int q = 0; q < 12; ++q) {
        int pp = pf[q];
        fr[q] = h[(size_t)(6 + q) * DD + d]          + (dr[pp] * er1 - di[pp] * ei1);
        fi[q] = h[(size_t)(6 + q) * DD + HALF_D + d] + (dr[pp] * ei1 + di[pp] * er1);
        x[(size_t)(6 + q) * DD + d]          = fr[q] * (1.0f / 3.0f);
        x[(size_t)(6 + q) * DD + HALF_D + d] = fi[q] * (1.0f / 3.0f);
    }
#pragma unroll
    for (int c = 0; c < 36; ++c) {
        int pp = pidx[c];
        float ir = h[(size_t)(18 + c) * DD + d]          + (fr[pp] * er2 - fi[pp] * ei2);
        float ii = h[(size_t)(18 + c) * DD + HALF_D + d] + (fr[pp] * ei2 + fi[pp] * er2);
        x[(size_t)(18 + c) * DD + d]          = ir * 0.25f;
        x[(size_t)(18 + c) * DD + HALF_D + d] = ii * 0.25f;
    }
}

__global__ void k_copy_out(const float* __restrict__ x, float* __restrict__ out) {
    int idx = blockIdx.x * blockDim.x + threadIdx.x;   // 54*4096
    out[idx] = x[idx];
}

// ---------------------------------------------------------------------------
// Host orchestration: 9 kernels per layer, 4 layers, all on `stream`.
// Workspace layout: x (1MB) | xh (.5MB) | xl (.5MB) | h (1MB) | hpart | coef
// ---------------------------------------------------------------------------
template <int SK>
static void run_pipeline(const float* ce, const float* gat_W,
                         const float* att_if, const float* att_fd,
                         const float* att_dr, const float* meta_W,
                         const float* edge, const int* pf, const int* pidx,
                         float* d_out, char* ws, hipStream_t stream) {
    float*  x     = (float*)(ws);                                   // 1 MB
    __bf16* xh    = (__bf16*)(ws + (1u << 20));                     // 512 KB
    __bf16* xl    = (__bf16*)(ws + (1u << 20) + (512u << 10));      // 512 KB
    float*  h     = (float*)(ws + (2u << 20));                      // 1 MB
    float*  hpart = (float*)(ws + (3u << 20));                      // SK MB
    float*  coef  = (float*)(ws + (3u << 20) + (size_t)SK * (1u << 20));

    k_init_x<<<1024, 256, 0, stream>>>(ce, x);
    for (int i = 0; i < LAYERS; ++i) {
        const float* Wg = gat_W  + (size_t)i * DD * DD;
        const float* Wm = meta_W + (size_t)i * DD * DD;

        k_prep<<<1024, 256, 0, stream>>>(x, xh, xl);
        k_gemm<SK><<<dim3(64, SK), 128, 0, stream>>>(Wg, xh, xl, hpart);
        k_reduce<SK><<<256, 256, 0, stream>>>(hpart, h);
        k_attn_coef<<<1, 1024, 0, stream>>>(h,
            att_if + (size_t)i * 2 * DD, att_fd + (size_t)i * 2 * DD,
            att_dr + (size_t)i * 2 * DD, pf, pidx, coef);
        k_attn_apply<<<864, 256, 0, stream>>>(h, coef, x);

        k_prep<<<1024, 256, 0, stream>>>(x, xh, xl);
        k_gemm<SK><<<dim3(64, SK), 128, 0, stream>>>(Wm, xh, xl, hpart);
        k_reduce<SK><<<256, 256, 0, stream>>>(hpart, h);
        k_mix<<<8, 256, 0, stream>>>(h, edge, pf, pidx, x);
    }
    k_copy_out<<<864, 256, 0, stream>>>(x, d_out);
}

extern "C" void kernel_launch(void* const* d_in, const int* in_sizes, int n_in,
                              void* d_out, int out_size, void* d_ws, size_t ws_size,
                              hipStream_t stream) {
    (void)in_sizes; (void)n_in; (void)out_size;
    const float* ce     = (const float*)d_in[0];
    const float* gat_W  = (const float*)d_in[1];
    const float* att_if = (const float*)d_in[2];
    const float* att_fd = (const float*)d_in[3];
    const float* att_dr = (const float*)d_in[4];
    const float* meta_W = (const float*)d_in[5];
    const float* edge   = (const float*)d_in[6];
    const int*   pf     = (const int*)d_in[7];
    const int*   pidx   = (const int*)d_in[8];
    char* ws = (char*)d_ws;

    // Pick split-K from available workspace: need 3MB + SK MB (+4KB coef).
    const size_t fixed = (3u << 20) + (64u << 10);
    if (ws_size == 0 || ws_size >= fixed + (16u << 20)) {
        run_pipeline<16>(ce, gat_W, att_if, att_fd, att_dr, meta_W, edge,
                         pf, pidx, (float*)d_out, ws, stream);
    } else if (ws_size >= fixed + (8u << 20)) {
        run_pipeline<8>(ce, gat_W, att_if, att_fd, att_dr, meta_W, edge,
                        pf, pidx, (float*)d_out, ws, stream);
    } else {
        run_pipeline<4>(ce, gat_W, att_if, att_fd, att_dr, meta_W, edge,
                        pf, pidx, (float*)d_out, ws, stream);
    }
}
